// MEAttention_83081847374478
// MI455X (gfx1250) — matrix operand order repaired and checked by reading, not compiled, not run on hardware
//
#include <hip/hip_runtime.h>

// ---------- types ----------
typedef __bf16        v16bf __attribute__((ext_vector_type(16)));
typedef float         v8f   __attribute__((ext_vector_type(8)));
typedef float         v4f   __attribute__((ext_vector_type(4)));
typedef unsigned int  v4u   __attribute__((ext_vector_type(4)));
typedef unsigned int  v2u   __attribute__((ext_vector_type(2)));

#define DEV static __device__ __forceinline__

DEV unsigned short f2bf(float f) {
  union { float f; unsigned u; } c; c.f = f;
  unsigned u = c.u;
  return (unsigned short)((u + 0x7fffu + ((u >> 16) & 1u)) >> 16);
}

union Frag { v16bf v; v4u u[2]; };

// problem constants
#define NB  32
#define CC_ 256
#define HH_ 56
#define NN_ 3136          // 56*56
#define K1_ 16384         // 256*8*8
#define K2_ 4096          // 256*4*4
#define M1R 1568          // 32*49
#define M2R 6272          // 32*196

// epilogues
#define EPI_F32R  0
#define EPI_F32T  1
#define EPI_BF16T 2
#define EPI_QSOFT 3

// =====================================================================
// Generic bf16 WMMA GEMM: S = A(MxK,row-major bf16) * Bt(256xK,row-major bf16)^T
// block = 256 threads (8 waves); BM = 32 rows; all 256 columns per block.
// Each wave: 2(m) x 2(c) accumulator tiles; B fragment reused across both
// m-tiles (halves global B traffic per WMMA). A tile double-buffered in LDS
// -> single workgroup barrier per 32-wide K step.
// =====================================================================
template <int EPI>
__global__ __launch_bounds__(256) void gemm_bf16_k(
    const unsigned short* __restrict__ A,  long lda, long strideA,
    const unsigned short* __restrict__ Bt, long ldb, long strideB,
    const float* __restrict__ bias,
    void* __restrict__ Out, long ldo, long strideO,
    int K)
{
  extern __shared__ char smem_raw[];
  unsigned short* Atile = (unsigned short*)smem_raw;      // 2 x (32 x 48) bf16

  const int tid  = threadIdx.x;
  const int lane = tid & 31;
  const int wid  = tid >> 5;        // 0..7 : column group (2 tiles of 16)
  const int b    = blockIdx.y;
  const long m0  = (long)blockIdx.x * 32;

  const unsigned short* Ab = A  + (long)b * strideA;
  const unsigned short* Bb = Bt + (long)b * strideB;

  v8f acc[2][2];
#pragma unroll
  for (int mi = 0; mi < 2; ++mi)
#pragma unroll
    for (int ci = 0; ci < 2; ++ci)
#pragma unroll
      for (int i = 0; i < 8; ++i) acc[mi][ci][i] = 0.0f;

  const int ar = tid >> 3;          // staging row 0..31
  const int ac = (tid & 7) * 4;     // staging col (4 bf16 per thread)
  const int fm = lane & 15;
  const int hi = lane >> 4;

  // preload first A tile into buffer 0
  {
    v2u av0 = *(const v2u*)(Ab + (m0 + ar) * lda + ac);
    *(v2u*)(Atile + ar * 48 + ac) = av0;
  }

  for (int k0 = 0; k0 < K; k0 += 32) {
    const int cur = (k0 >> 5) & 1;
    __syncthreads();                                   // publish tile `cur`

    v2u avn;
    const bool more = (k0 + 32) < K;
    if (more) {
      avn = *(const v2u*)(Ab + (m0 + ar) * lda + k0 + 32 + ac);
      if (k0 + 64 < K)                                 // gfx1250 prefetch path
        __builtin_prefetch(Ab + (m0 + ar) * lda + k0 + 64 + ac, 0, 3);
    }

    // A fragments per ISA 7.12.2 (16-bit A 16x32):
    // lane hi=0 -> K 0..7,16..23 ; hi=1 -> K 8..15,24..31
    const unsigned short* base = Atile + cur * 1536;
    const unsigned short* ap0 = base + fm * 48 + hi * 8;
    const unsigned short* ap1 = base + (16 + fm) * 48 + hi * 8;
    Frag fa0, fa1;
    fa0.u[0] = *(const v4u*)ap0; fa0.u[1] = *(const v4u*)(ap0 + 16);
    fa1.u[0] = *(const v4u*)ap1; fa1.u[1] = *(const v4u*)(ap1 + 16);

#pragma unroll
    for (int ci = 0; ci < 2; ++ci) {
      const int cc = (wid * 2 + ci) * 16 + fm;         // B column = lane%16
      const unsigned short* bp = Bb + (long)cc * ldb + k0 + hi * 8; // K 8h..8h+7, 16+8h..23+8h
      Frag fb; fb.u[0] = *(const v4u*)bp; fb.u[1] = *(const v4u*)(bp + 16);
      acc[0][ci] = __builtin_amdgcn_wmma_f32_16x16x32_bf16(
          false, fa0.v, false, fb.v, (short)0, acc[0][ci], false, false);
      acc[1][ci] = __builtin_amdgcn_wmma_f32_16x16x32_bf16(
          false, fa1.v, false, fb.v, (short)0, acc[1][ci], false, false);
    }

    if (more)                                          // stage into other buffer
      *(v2u*)(Atile + (cur ^ 1) * 1536 + ar * 48 + ac) = avn;
  }

  const int mb = hi * 8;                               // D layout: rows mb..mb+7, col = lane%16

  if constexpr (EPI == EPI_F32R) {
    float* O = (float*)Out + (long)b * strideO;
#pragma unroll
    for (int mi = 0; mi < 2; ++mi) {
      const long r0 = m0 + mi * 16 + mb;
#pragma unroll
      for (int ci = 0; ci < 2; ++ci) {
        const int cc = (wid * 2 + ci) * 16 + fm;
        const float bv = bias ? bias[cc] : 0.0f;
#pragma unroll
        for (int r = 0; r < 8; ++r) O[(r0 + r) * ldo + cc] = acc[mi][ci][r] + bv;
      }
    }
  } else if constexpr (EPI == EPI_F32T) {
    float* O = (float*)Out + (long)b * strideO;
#pragma unroll
    for (int mi = 0; mi < 2; ++mi) {
      const long r0 = m0 + mi * 16 + mb;
#pragma unroll
      for (int ci = 0; ci < 2; ++ci) {
        const int cc = (wid * 2 + ci) * 16 + fm;
        const float bv = bias ? bias[cc] : 0.0f;
        float* p = O + (long)cc * ldo + r0;
        v4f lo, hi4;
#pragma unroll
        for (int r = 0; r < 4; ++r) { lo[r] = acc[mi][ci][r] + bv; hi4[r] = acc[mi][ci][4 + r] + bv; }
        *(v4f*)p = lo; *(v4f*)(p + 4) = hi4;
      }
    }
  } else if constexpr (EPI == EPI_BF16T) {
    unsigned short* O = (unsigned short*)Out + (long)b * strideO;
#pragma unroll
    for (int mi = 0; mi < 2; ++mi) {
      const long r0 = m0 + mi * 16 + mb;
#pragma unroll
      for (int ci = 0; ci < 2; ++ci) {
        const int cc = (wid * 2 + ci) * 16 + fm;
        const float bv = bias ? bias[cc] : 0.0f;
        union { unsigned short h[8]; v4u u; } pk;
#pragma unroll
        for (int r = 0; r < 8; ++r) pk.h[r] = f2bf(acc[mi][ci][r] + bv);
        *(v4u*)(O + (long)cc * ldo + r0) = pk.u;
      }
    }
  } else {  // EPI_QSOFT : row softmax over 256 columns, bf16 row-major out
    float* sS   = (float*)(smem_raw + 3072);           // 32 x 260 (padded)
    float* pbuf = sS + 32 * 260;                       // 256 partials
    float* rred = pbuf + 256;                          // 32 row values
#pragma unroll
    for (int mi = 0; mi < 2; ++mi)
#pragma unroll
      for (int ci = 0; ci < 2; ++ci) {
        const int cc = (wid * 2 + ci) * 16 + fm;
        const float bv = bias[cc];
#pragma unroll
        for (int r = 0; r < 8; ++r) sS[(mi * 16 + mb + r) * 260 + cc] = acc[mi][ci][r] + bv;
      }
    __syncthreads();
    const int row = tid >> 3, seg = tid & 7;
    float mx = -3.4e38f;
#pragma unroll
    for (int j = 0; j < 32; ++j) mx = fmaxf(mx, sS[row * 260 + seg * 32 + j]);
    pbuf[tid] = mx; __syncthreads();
    if (tid < 32) {
      float m2 = pbuf[tid * 8];
#pragma unroll
      for (int j = 1; j < 8; ++j) m2 = fmaxf(m2, pbuf[tid * 8 + j]);
      rred[tid] = m2;
    }
    __syncthreads();
    const float rm = rred[row];
    float sum = 0.0f;
#pragma unroll
    for (int j = 0; j < 32; ++j) sum += expf(sS[row * 260 + seg * 32 + j] - rm);
    __syncthreads();
    pbuf[tid] = sum; __syncthreads();
    if (tid < 32) {
      float s2 = 0.0f;
#pragma unroll
      for (int j = 0; j < 8; ++j) s2 += pbuf[tid * 8 + j];
      rred[tid] = s2;
    }
    __syncthreads();
    const float inv = 1.0f / rred[row];
    unsigned short* O = (unsigned short*)Out + (long)b * strideO + (m0 + row) * ldo;
#pragma unroll
    for (int j = 0; j < 32; ++j) {
      const int cc = seg * 32 + j;
      O[cc] = f2bf(expf(sS[row * 260 + cc] - rm) * inv);
    }
  }
}

// ------------- transpose + f32->bf16 pack: dst[c][r] = src[r][c] -------------
__global__ __launch_bounds__(256) void transpose_cvt_k(
    const float* __restrict__ src, unsigned short* __restrict__ dst,
    int R, int C, long sstride, long dstride)
{
  __shared__ float tile[32][33];
  const int b  = blockIdx.z;
  const float* s = src + (long)b * sstride;
  unsigned short* d = dst + (long)b * dstride;
  const int c0 = blockIdx.x * 32, r0 = blockIdx.y * 32;
  const int tx = threadIdx.x, ty = threadIdx.y;       // 32 x 8
#pragma unroll
  for (int i = 0; i < 4; ++i)
    tile[ty + i * 8][tx] = s[(long)(r0 + ty + i * 8) * C + c0 + tx];
  __syncthreads();
#pragma unroll
  for (int i = 0; i < 4; ++i)
    d[(long)(c0 + ty + i * 8) * R + r0 + tx] = f2bf(tile[tx][ty + i * 8]);
}

// ------------- straight f32 -> bf16 convert -------------
__global__ __launch_bounds__(256) void cvt_k(const float* __restrict__ s,
                                             unsigned short* __restrict__ d, long n)
{
  long i = (long)blockIdx.x * 256 + threadIdx.x;
  if (i < n) d[i] = f2bf(s[i]);
}

// ------------- im2col for stride==kernel patch conv (pure permute) -------------
__global__ __launch_bounds__(256) void im2col_k(const float* __restrict__ x,
                                                unsigned short* __restrict__ A,
                                                int S, int OH, int K)
{
  long idx = (long)blockIdx.x * 256 + threadIdx.x;
  int k   = (int)(idx % K);
  long row = idx / K;
  int c  = k / (S * S), rr = k % (S * S), ky = rr / S, kx = rr % S;
  int b  = (int)(row / (OH * OH)); int p = (int)(row % (OH * OH));
  int oy = p / OH, ox = p % OH;
  float v = x[(((long)b * CC_ + c) * HH_ + oy * S + ky) * HH_ + ox * S + kx];
  A[idx] = f2bf(v);
}

// ------------- column softmax (over N) of sKT[b][c][:] -> bf16 -------------
__global__ __launch_bounds__(256) void col_softmax_k(const float* __restrict__ S,
                                                     unsigned short* __restrict__ O, int n)
{
  __shared__ float red[256];
  const long base = (long)blockIdx.x * n;
  const int tid = threadIdx.x;
  float mx = -3.4e38f;
  for (int i = tid; i < n; i += 256) mx = fmaxf(mx, S[base + i]);
  red[tid] = mx; __syncthreads();
  for (int s = 128; s > 0; s >>= 1) { if (tid < s) red[tid] = fmaxf(red[tid], red[tid + s]); __syncthreads(); }
  const float m = red[0]; __syncthreads();
  float sum = 0.0f;
  for (int i = tid; i < n; i += 256) sum += expf(S[base + i] - m);
  red[tid] = sum; __syncthreads();
  for (int s = 128; s > 0; s >>= 1) { if (tid < s) red[tid] += red[tid + s]; __syncthreads(); }
  const float inv = 1.0f / red[0];
  for (int i = tid; i < n; i += 256) O[base + i] = f2bf(expf(S[base + i] - m) * inv);
}

// ------------- LayerNorm(256) + exact GELU, f32 -> bf16 -------------
__global__ __launch_bounds__(256) void ln_gelu_k(const float* __restrict__ S,
                                                 const float* __restrict__ g,
                                                 const float* __restrict__ bb,
                                                 unsigned short* __restrict__ O)
{
  __shared__ float red[256];
  const long row = blockIdx.x; const int c = threadIdx.x;
  const float v = S[row * 256 + c];
  red[c] = v; __syncthreads();
  for (int s = 128; s > 0; s >>= 1) { if (c < s) red[c] += red[c + s]; __syncthreads(); }
  const float mu = red[0] * (1.0f / 256.0f); __syncthreads();
  const float d = v - mu;
  red[c] = d * d; __syncthreads();
  for (int s = 128; s > 0; s >>= 1) { if (c < s) red[c] += red[c + s]; __syncthreads(); }
  const float var = red[0] * (1.0f / 256.0f);
  const float y = d * rsqrtf(var + 1e-5f) * g[c] + bb[c];
  const float ge = 0.5f * y * (1.0f + erff(y * 0.70710678118654752f));
  O[row * 256 + c] = f2bf(ge);
}

// ------------- branch prep: k-softmax over tokens + depthwise 3x3 on v -------------
template <int NN, int HHG>
__global__ __launch_bounds__(256) void branch_prep_k(const float* __restrict__ kv,
                                                     const float* __restrict__ lcw,
                                                     const float* __restrict__ lcb,
                                                     float* __restrict__ kS,
                                                     float* __restrict__ vP)
{
  const int b = blockIdx.x, tid = threadIdx.x;
  const float* kvb = kv + (long)b * NN * 256;
  if (tid < 128) {
    const int ch = tid;
    float m = -3.4e38f;
    for (int n = 0; n < NN; ++n) m = fmaxf(m, kvb[n * 256 + ch]);
    float s = 0.0f;
    for (int n = 0; n < NN; ++n) s += expf(kvb[n * 256 + ch] - m);
    const float inv = 1.0f / s;
    for (int n = 0; n < NN; ++n)
      kS[((long)b * NN + n) * 128 + ch] = expf(kvb[n * 256 + ch] - m) * inv;
  }
  for (int e = tid; e < NN * 128; e += 256) {
    const int n = e >> 7, ch = e & 127;
    const int y = n / HHG, x = n % HHG;
    float a = lcb[ch];
#pragma unroll
    for (int ky = 0; ky < 3; ++ky)
#pragma unroll
      for (int kx = 0; kx < 3; ++kx) {
        const int yy = y + ky - 1, xx = x + kx - 1;
        if (yy >= 0 && yy < HHG && xx >= 0 && xx < HHG)
          a += lcw[ch * 9 + ky * 3 + kx] * kvb[(yy * HHG + xx) * 256 + 128 + ch];
      }
    vP[((long)b * NN + n) * 128 + ch] = kvb[n * 256 + 128 + ch] + a;
  }
}

// ------------- branch ctx: ctx[b][kc][vc] = sum_n kS * vP  (128x128, K=NN) -------------
template <int NN>
__global__ __launch_bounds__(256) void branch_ctx_k(const float* __restrict__ kS,
                                                    const float* __restrict__ vP,
                                                    float* __restrict__ ctx)
{
  const int b = blockIdx.x;
  const int e = blockIdx.y * 256 + threadIdx.x;       // 0..16383
  const int kc = e >> 7, vc = e & 127;
  const float* kb = kS + (long)b * NN * 128;
  const float* vb = vP + (long)b * NN * 128;
  float s = 0.0f;
  for (int n = 0; n < NN; ++n) s += kb[n * 128 + kc] * vb[n * 128 + vc];
  ctx[(long)b * 16384 + e] = s;
}

// ------------- build Mfull^T (bf16) and constant bias vector -------------
__global__ __launch_bounds__(256) void mfull_k(const float* __restrict__ ctx,
                                               const float* __restrict__ ctx1,
                                               const float* __restrict__ ctx2,
                                               const float* __restrict__ rpw,
                                               const float* __restrict__ rp12w,
                                               const float* __restrict__ rpb,
                                               const float* __restrict__ rp12b,
                                               const float* __restrict__ dww,
                                               unsigned short* __restrict__ mT,
                                               float* __restrict__ cvec)
{
  const int b = blockIdx.y;
  const int o = blockIdx.x * 32 + (threadIdx.x & 31);
  const int kg = threadIdx.x >> 5;
  const float dw0 = dww[o * 2], dw1 = dww[o * 2 + 1];
  if (b == 0 && kg == 0) cvec[o] = dw0 * rpb[o] + dw1 * rp12b[o];
  for (int k = kg * 32; k < kg * 32 + 32; ++k) {
    float dA = 0.0f;
    const float* cr = ctx + ((long)b * 256 + k) * 256;
    const float* wr = rpw + (long)o * 256;
    for (int i = 0; i < 256; ++i) dA += cr[i] * wr[i];
    float dB = 0.0f;
    if (k < 128) {
      const float* c1 = ctx1 + ((long)b * 128 + k) * 128;
      const float* w12 = rp12w + (long)o * 256;
      for (int i = 0; i < 128; ++i) dB += c1[i] * w12[i];
    } else {
      const float* c2 = ctx2 + ((long)b * 128 + (k - 128)) * 128;
      const float* w12 = rp12w + (long)o * 256 + 128;
      for (int i = 0; i < 128; ++i) dB += c2[i] * w12[i];
    }
    mT[((long)b * 256 + o) * 256 + k] = f2bf(dw0 * dA + dw1 * dB);
  }
}

// =====================================================================
extern "C" void kernel_launch(void* const* d_in, const int* in_sizes, int n_in,
                              void* d_out, int out_size, void* d_ws, size_t ws_size,
                              hipStream_t stream)
{
  (void)in_sizes; (void)n_in; (void)out_size; (void)ws_size;
  const float* x     = (const float*)d_in[0];
  const float* Wq    = (const float*)d_in[1];
  const float* bq    = (const float*)d_in[2];
  const float* Wk    = (const float*)d_in[3];
  const float* bk    = (const float*)d_in[4];
  const float* Wv    = (const float*)d_in[5];
  const float* bv    = (const float*)d_in[6];
  const float* sr1w  = (const float*)d_in[7];
  const float* sr1b  = (const float*)d_in[8];
  const float* ln1g  = (const float*)d_in[9];
  const float* ln1b  = (const float*)d_in[10];
  const float* sr2w  = (const float*)d_in[11];
  const float* sr2b  = (const float*)d_in[12];
  const float* ln2g  = (const float*)d_in[13];
  const float* ln2b  = (const float*)d_in[14];
  const float* Wkv1  = (const float*)d_in[15];
  const float* bkv1  = (const float*)d_in[16];
  const float* Wkv2  = (const float*)d_in[17];
  const float* bkv2  = (const float*)d_in[18];
  const float* lc1w  = (const float*)d_in[19];
  const float* lc1b  = (const float*)d_in[20];
  const float* lc2w  = (const float*)d_in[21];
  const float* lc2b  = (const float*)d_in[22];
  const float* rpw   = (const float*)d_in[23];
  const float* rpb   = (const float*)d_in[24];
  const float* rp12w = (const float*)d_in[25];
  const float* rp12b = (const float*)d_in[26];
  const float* dww   = (const float*)d_in[27];
  float* out = (float*)d_out;

  // ---- workspace layout ----
  char* W = (char*)d_ws;
  size_t off = 0;
  auto al = [&](size_t bytes) { size_t o = off; off += (bytes + 255) & ~(size_t)255; return o; };
  const size_t o_xt   = al((size_t)NB * NN_ * CC_ * 2);
  const size_t o_wqT  = al((size_t)CC_ * CC_ * 2);
  const size_t o_wkT  = al((size_t)CC_ * CC_ * 2);
  const size_t o_wvT  = al((size_t)CC_ * CC_ * 2);
  const size_t o_wk1T = al((size_t)CC_ * CC_ * 2);
  const size_t o_wk2T = al((size_t)CC_ * CC_ * 2);
  const size_t o_s1T  = al((size_t)CC_ * K1_ * 2);
  const size_t o_s2T  = al((size_t)CC_ * K2_ * 2);
  const size_t o_qh   = al((size_t)NB * NN_ * CC_ * 2);
  const size_t o_sKT  = al((size_t)NB * CC_ * NN_ * 4);   // later reused for im2col A1|A2
  const size_t o_kT   = al((size_t)NB * CC_ * NN_ * 2);
  const size_t o_vT   = al((size_t)NB * CC_ * NN_ * 2);
  const size_t o_ctx  = al((size_t)NB * CC_ * CC_ * 4);
  const size_t o_x1s  = al((size_t)M1R * CC_ * 4);
  const size_t o_x2s  = al((size_t)M2R * CC_ * 4);
  const size_t o_x1h  = al((size_t)M1R * CC_ * 2);
  const size_t o_x2h  = al((size_t)M2R * CC_ * 2);
  const size_t o_kv1  = al((size_t)M1R * CC_ * 4);
  const size_t o_kv2  = al((size_t)M2R * CC_ * 4);
  const size_t o_kS1  = al((size_t)M1R * 128 * 4);
  const size_t o_vP1  = al((size_t)M1R * 128 * 4);
  const size_t o_kS2  = al((size_t)M2R * 128 * 4);
  const size_t o_vP2  = al((size_t)M2R * 128 * 4);
  const size_t o_c1   = al((size_t)NB * 128 * 128 * 4);
  const size_t o_c2   = al((size_t)NB * 128 * 128 * 4);
  const size_t o_mT   = al((size_t)NB * CC_ * CC_ * 2);
  const size_t o_cv   = al((size_t)256 * 4);

  unsigned short* xt  = (unsigned short*)(W + o_xt);
  unsigned short* wqT = (unsigned short*)(W + o_wqT);
  unsigned short* wkT = (unsigned short*)(W + o_wkT);
  unsigned short* wvT = (unsigned short*)(W + o_wvT);
  unsigned short* wk1T= (unsigned short*)(W + o_wk1T);
  unsigned short* wk2T= (unsigned short*)(W + o_wk2T);
  unsigned short* s1T = (unsigned short*)(W + o_s1T);
  unsigned short* s2T = (unsigned short*)(W + o_s2T);
  unsigned short* qh  = (unsigned short*)(W + o_qh);
  float*          sKT = (float*)(W + o_sKT);
  unsigned short* A1  = (unsigned short*)(W + o_sKT);                         // reuse
  unsigned short* A2  = (unsigned short*)(W + o_sKT + (size_t)M1R * K1_ * 2); // reuse
  unsigned short* kT  = (unsigned short*)(W + o_kT);
  unsigned short* vT  = (unsigned short*)(W + o_vT);
  float* ctxf = (float*)(W + o_ctx);
  float* x1s  = (float*)(W + o_x1s);
  float* x2s  = (float*)(W + o_x2s);
  unsigned short* x1h = (unsigned short*)(W + o_x1h);
  unsigned short* x2h = (unsigned short*)(W + o_x2h);
  float* kv1f = (float*)(W + o_kv1);
  float* kv2f = (float*)(W + o_kv2);
  float* kS1  = (float*)(W + o_kS1);
  float* vP1  = (float*)(W + o_vP1);
  float* kS2  = (float*)(W + o_kS2);
  float* vP2  = (float*)(W + o_vP2);
  float* c1f  = (float*)(W + o_c1);
  float* c2f  = (float*)(W + o_c2);
  unsigned short* mT = (unsigned short*)(W + o_mT);
  float* cvec = (float*)(W + o_cv);

  const long NC = (long)NN_ * CC_;       // per-batch elems of (N,C)
  const long CN = (long)CC_ * NN_;
  const int SM_G = 3072;                 // double-buffered Atile
  const int SM_Q = 3072 + 32 * 260 * 4 + 256 * 4 + 32 * 4;

  // ---- pack operands to bf16 ----
  transpose_cvt_k<<<dim3(NN_ / 32, CC_ / 32, NB), dim3(32, 8), 0, stream>>>(
      x, xt, CC_, NN_, (long)CC_ * NN_, NC);
  transpose_cvt_k<<<dim3(8, 8, 1), dim3(32, 8), 0, stream>>>(Wq,   wqT,  256, 256, 0, 0);
  transpose_cvt_k<<<dim3(8, 8, 1), dim3(32, 8), 0, stream>>>(Wk,   wkT,  256, 256, 0, 0);
  transpose_cvt_k<<<dim3(8, 8, 1), dim3(32, 8), 0, stream>>>(Wv,   wvT,  256, 256, 0, 0);
  transpose_cvt_k<<<dim3(8, 8, 1), dim3(32, 8), 0, stream>>>(Wkv1, wk1T, 256, 256, 0, 0);
  transpose_cvt_k<<<dim3(8, 8, 1), dim3(32, 8), 0, stream>>>(Wkv2, wk2T, 256, 256, 0, 0);
  cvt_k<<<(int)(((long)CC_ * K1_ + 255) / 256), 256, 0, stream>>>(sr1w, s1T, (long)CC_ * K1_);
  cvt_k<<<(int)(((long)CC_ * K2_ + 255) / 256), 256, 0, stream>>>(sr2w, s2T, (long)CC_ * K2_);

  // ---- q / k / v projections (per-batch grid.y) ----
  gemm_bf16_k<EPI_QSOFT><<<dim3(98, NB), 256, SM_Q, stream>>>(
      xt, 256, NC, wqT, 256, 0, bq, qh, 256, NC, 256);
  gemm_bf16_k<EPI_F32T><<<dim3(98, NB), 256, SM_G, stream>>>(
      xt, 256, NC, wkT, 256, 0, bk, sKT, NN_, CN, 256);
  gemm_bf16_k<EPI_BF16T><<<dim3(98, NB), 256, SM_G, stream>>>(
      xt, 256, NC, wvT, 256, 0, bv, vT, NN_, CN, 256);
  col_softmax_k<<<NB * CC_, 256, 0, stream>>>(sKT, kT, NN_);

  // ---- big attention context: ctx[b] = keys^T @ values  (K = 3136) ----
  gemm_bf16_k<EPI_F32R><<<dim3(8, NB), 256, SM_G, stream>>>(
      kT, NN_, CN, vT, NN_, CN, nullptr, ctxf, 256, (long)CC_ * CC_, NN_);

  // ---- patch convs as GEMMs (stride==kernel -> im2col is a permute) ----
  im2col_k<<<(int)(((long)M1R * K1_) / 256), 256, 0, stream>>>(x, A1, 8, 7, K1_);
  gemm_bf16_k<EPI_F32R><<<dim3(M1R / 32, 1), 256, SM_G, stream>>>(
      A1, K1_, 0, s1T, K1_, 0, sr1b, x1s, 256, 0, K1_);
  ln_gelu_k<<<M1R, 256, 0, stream>>>(x1s, ln1g, ln1b, x1h);

  im2col_k<<<(int)(((long)M2R * K2_) / 256), 256, 0, stream>>>(x, A2, 4, 14, K2_);
  gemm_bf16_k<EPI_F32R><<<dim3(M2R / 32, 1), 256, SM_G, stream>>>(
      A2, K2_, 0, s2T, K2_, 0, sr2b, x2s, 256, 0, K2_);
  ln_gelu_k<<<M2R, 256, 0, stream>>>(x2s, ln2g, ln2b, x2h);

  // ---- kv projections + branch contexts ----
  gemm_bf16_k<EPI_F32R><<<dim3(M1R / 32, 1), 256, SM_G, stream>>>(
      x1h, 256, 0, wk1T, 256, 0, bkv1, kv1f, 256, 0, 256);
  gemm_bf16_k<EPI_F32R><<<dim3(M2R / 32, 1), 256, SM_G, stream>>>(
      x2h, 256, 0, wk2T, 256, 0, bkv2, kv2f, 256, 0, 256);
  branch_prep_k<49, 7><<<NB, 256, 0, stream>>>(kv1f, lc1w, lc1b, kS1, vP1);
  branch_prep_k<196, 14><<<NB, 256, 0, stream>>>(kv2f, lc2w, lc2b, kS2, vP2);
  branch_ctx_k<49><<<dim3(NB, 64), 256, 0, stream>>>(kS1, vP1, c1f);
  branch_ctx_k<196><<<dim3(NB, 64), 256, 0, stream>>>(kS2, vP2, c2f);

  // ---- fold ctx / ctx1 / ctx2 / rp / rp12 / dw into one per-batch 256x256 matrix ----
  mfull_k<<<dim3(8, NB), 256, 0, stream>>>(ctxf, c1f, c2f, rpw, rp12w, rpb, rp12b, dww, mT, cvec);

  // ---- final: out[b,o,n] = q[b,n,:] . Mfull[b,:,o] + cvec[o]  (transposed store -> NCHW) ----
  gemm_bf16_k<EPI_F32T><<<dim3(98, NB), 256, SM_G, stream>>>(
      qh, 256, NC, mT, 256, (long)CC_ * CC_, cvec, out, NN_, CN, 256);
}
